// MyLossJC_71236327571663
// MI455X (gfx1250) — matrix operand order repaired and checked by reading, not compiled, not run on hardware
//
#include <hip/hip_runtime.h>
#include <hip/hip_bf16.h>

typedef __attribute__((ext_vector_type(16))) _Float16 v16h;
typedef __attribute__((ext_vector_type(8)))  float    v8f;
typedef __attribute__((ext_vector_type(4)))  float    v4f;   // native vector for NT loads

#define NBATCH 8
#define NCLS   19
#define HWPIX  (512 * 512)

// 32x32 padded S per batch (row 19 of A is all-ones => S[19][k] = counts[k])
#define S_STRIDE 1024           // 32*32

#define WG_PER_BATCH 64
#define NWG          (NBATCH * WG_PER_BATCH)          // 512
#define PIX_PER_WG   (HWPIX / WG_PER_BATCH)           // 4096
#define WAVES_PER_WG 8
#define PIX_PER_WAVE (PIX_PER_WG / WAVES_PER_WG)      // 512
#define ITERS        (PIX_PER_WAVE / 32)              // 16

// workspace layout: [NWG][1024] partial tiles, then [NBATCH][1024] final tiles
#define WS_PARTIALS  ((size_t)NWG * S_STRIDE)         // 512K floats = 2 MB
#define WS_FINAL_OFF WS_PARTIALS

union H16 { v16h v; _Float16 h[16]; };

__global__ __launch_bounds__(256)
void seg_gemm_kernel(const float* __restrict__ Z,       // [B, C, H, W] f32
                     const long long* __restrict__ T,   // [B, H, W] i64
                     float* __restrict__ P)             // [NWG, 32, 32] partials
{
    __shared__ float sAcc[WAVES_PER_WG][S_STRIDE];      // 32 KB

    const int tid  = threadIdx.x;
    const int wg   = blockIdx.x;
    const int b    = wg / WG_PER_BATCH;
    const int chnk = wg % WG_PER_BATCH;
    const int wave = tid >> 5;
    const int lane = tid & 31;

    const float*     Zb = Z + (size_t)b * NCLS * HWPIX;
    const long long* Tb = T + (size_t)b * HWPIX;
    const int pixBase = chnk * PIX_PER_WG + wave * PIX_PER_WAVE;

    // CDNA5 16-bit A layout (16x32): lanes L and L+16 hold row M=L&15.
    // lanes 0-15: halves 0..7 <-> K=0..7,  halves 8..15 <-> K=16..23
    // lanes 16-31: halves 0..7 <-> K=8..15, halves 8..15 <-> K=24..31
    const int col = lane & 15;             // A row / B column index
    const int ka  = (lane & 16) ? 8 : 0;   // A pixel sub-offset
    const int kb  = (lane & 16) ? 8 : 0;   // B pixel sub-offset (same K split as A)
    const int c1  = col + 16;              // tile-1 class index (16..31)

    const float* rowA0 = Zb + (size_t)col * HWPIX;
    const float* rowA1 = Zb + (size_t)((c1 < NCLS) ? c1 : 0) * HWPIX;
    const bool      realA1 = (c1 < NCLS);
    const _Float16  padA1  = (c1 == NCLS) ? (_Float16)1.0f : (_Float16)0.0f; // ones row => counts

    v8f acc00 = {}, acc01 = {}, acc10 = {}, acc11 = {};

    for (int it = 0; it < ITERS; ++it) {
        const int p0 = pixBase + it * 32;

        // ---- A tile 0 (classes 0..15): two runs of 8 f32 (+0, +16) -> f16, streamed (NT) ----
        H16 a0;
        {
            const v4f* p = (const v4f*)(rowA0 + p0 + ka);
            v4f f0 = __builtin_nontemporal_load(p + 0);
            v4f f1 = __builtin_nontemporal_load(p + 1);
            v4f f2 = __builtin_nontemporal_load(p + 4);   // +16 floats
            v4f f3 = __builtin_nontemporal_load(p + 5);
            a0.h[0]=(_Float16)f0.x;  a0.h[1]=(_Float16)f0.y;  a0.h[2]=(_Float16)f0.z;  a0.h[3]=(_Float16)f0.w;
            a0.h[4]=(_Float16)f1.x;  a0.h[5]=(_Float16)f1.y;  a0.h[6]=(_Float16)f1.z;  a0.h[7]=(_Float16)f1.w;
            a0.h[8]=(_Float16)f2.x;  a0.h[9]=(_Float16)f2.y;  a0.h[10]=(_Float16)f2.z; a0.h[11]=(_Float16)f2.w;
            a0.h[12]=(_Float16)f3.x; a0.h[13]=(_Float16)f3.y; a0.h[14]=(_Float16)f3.z; a0.h[15]=(_Float16)f3.w;
        }

        // ---- A tile 1 (classes 16..18 real, 19 = ones row, 20..31 zero) ----
        H16 a1;
        {
            const v4f* p = (const v4f*)(rowA1 + p0 + ka);
            v4f f0 = __builtin_nontemporal_load(p + 0);
            v4f f1 = __builtin_nontemporal_load(p + 1);
            v4f f2 = __builtin_nontemporal_load(p + 4);
            v4f f3 = __builtin_nontemporal_load(p + 5);
            H16 tmp;
            tmp.h[0]=(_Float16)f0.x;  tmp.h[1]=(_Float16)f0.y;  tmp.h[2]=(_Float16)f0.z;  tmp.h[3]=(_Float16)f0.w;
            tmp.h[4]=(_Float16)f1.x;  tmp.h[5]=(_Float16)f1.y;  tmp.h[6]=(_Float16)f1.z;  tmp.h[7]=(_Float16)f1.w;
            tmp.h[8]=(_Float16)f2.x;  tmp.h[9]=(_Float16)f2.y;  tmp.h[10]=(_Float16)f2.z; tmp.h[11]=(_Float16)f2.w;
            tmp.h[12]=(_Float16)f3.x; tmp.h[13]=(_Float16)f3.y; tmp.h[14]=(_Float16)f3.z; tmp.h[15]=(_Float16)f3.w;
            #pragma unroll
            for (int j = 0; j < 16; ++j) a1.h[j] = realA1 ? tmp.h[j] : padA1;
        }

        // ---- B tiles: onehot(label == column), lane = column, K split by lane half ----
        int lab[16];
        {
            const long long* tp = Tb + p0 + kb;
            #pragma unroll
            for (int j = 0; j < 16; ++j) lab[j] = (int)tp[(j < 8) ? j : (j + 8)];
        }
        H16 b0, b1;
        #pragma unroll
        for (int j = 0; j < 16; ++j) {
            b0.h[j] = (lab[j] == col) ? (_Float16)1.0f : (_Float16)0.0f;
            b1.h[j] = (lab[j] == c1)  ? (_Float16)1.0f : (_Float16)0.0f;
        }

        // ---- 4 WMMAs: 32x32 accumulator tile, K += 32 ----
        acc00 = __builtin_amdgcn_wmma_f32_16x16x32_f16(false, a0.v, false, b0.v, (short)0, acc00, false, false);
        acc01 = __builtin_amdgcn_wmma_f32_16x16x32_f16(false, a0.v, false, b1.v, (short)0, acc01, false, false);
        acc10 = __builtin_amdgcn_wmma_f32_16x16x32_f16(false, a1.v, false, b0.v, (short)0, acc10, false, false);
        acc11 = __builtin_amdgcn_wmma_f32_16x16x32_f16(false, a1.v, false, b1.v, (short)0, acc11, false, false);
    }

    // ---- deterministic cross-wave reduction (no atomics) ----
    // C/D layout: lane 0-15: vgpr r = (M=r, N=lane); lane 16-31: vgpr r = (M=8+r, N=lane-16)
    {
        float* my = sAcc[wave];
        const int mBase = (lane >> 4) * 8;
        const int n     = lane & 15;
        #pragma unroll
        for (int r = 0; r < 8; ++r) {
            my[(mBase + r) * 32 + n]             = acc00[r];
            my[(mBase + r) * 32 + (n + 16)]      = acc01[r];
            my[(16 + mBase + r) * 32 + n]        = acc10[r];
            my[(16 + mBase + r) * 32 + (n + 16)] = acc11[r];
        }
    }
    __syncthreads();

    float* Pwg = P + (size_t)wg * S_STRIDE;
    for (int i = tid; i < S_STRIDE; i += 256) {
        float s = 0.0f;
        #pragma unroll
        for (int w = 0; w < WAVES_PER_WG; ++w) s += sAcc[w][i];
        Pwg[i] = s;   // fixed-order sum: bitwise deterministic
    }
}

__global__ __launch_bounds__(256)
void reduce_kernel(const float* __restrict__ P, float* __restrict__ S)
{
    int i = blockIdx.x * blockDim.x + threadIdx.x;      // 0 .. 8191
    if (i >= NBATCH * S_STRIDE) return;
    const int b = i >> 10;
    const int e = i & (S_STRIDE - 1);
    const float* base = P + (size_t)b * WG_PER_BATCH * S_STRIDE + e;
    float s = 0.0f;
    for (int w = 0; w < WG_PER_BATCH; ++w) s += base[(size_t)w * S_STRIDE];
    S[i] = s;
}

__global__ __launch_bounds__(256)
void finalize_kernel(const float* __restrict__ S, float* __restrict__ out)
{
    __shared__ float red[256];
    const int tid = threadIdx.x;
    const float eps = 2.2204460492503131e-16f;  // np.spacing(1)

    float sum = 0.0f;
    for (int idx = tid; idx < NBATCH * NCLS * NCLS; idx += 256) {
        int b   = idx / (NCLS * NCLS);
        int rem = idx % (NCLS * NCLS);
        int i   = rem / NCLS;
        int k   = rem % NCLS;
        const float* Sb = S + (size_t)b * S_STRIDE;
        float ck = Sb[19 * 32 + k];             // counts[b,k] from ones-row
        float ci = Sb[19 * 32 + i];             // counts[b,i]
        float invk = (ck > 0.0f) ? (1.0f / ck) : 0.0f;
        float invi = (ci > 0.0f) ? (1.0f / ci) : 0.0f;
        float alpha = invi * Sb[i * 32 + i];
        float beta  = invk * (ck - Sb[i * 32 + k]);
        sum += __logf(0.5f * (alpha + beta + eps));
    }
    red[tid] = sum;
    __syncthreads();
    for (int s = 128; s > 0; s >>= 1) {
        if (tid < s) red[tid] += red[tid + s];
        __syncthreads();
    }
    if (tid == 0) out[0] = -0.5f * red[0] / (float)NBATCH;
}

extern "C" void kernel_launch(void* const* d_in, const int* in_sizes, int n_in,
                              void* d_out, int out_size, void* d_ws, size_t ws_size,
                              hipStream_t stream) {
    (void)in_sizes; (void)n_in; (void)out_size; (void)ws_size;
    const float*     Z = (const float*)d_in[0];      // seg_softmax f32 [8,19,512,512]
    const long long* T = (const long long*)d_in[1];  // target int64 [8,512,512]
    float* P   = (float*)d_ws;                       // [512][1024] partial tiles (2 MB)
    float* S   = P + WS_FINAL_OFF;                   // [8][1024] final tiles (32 KB)
    float* out = (float*)d_out;                      // scalar f32

    seg_gemm_kernel<<<NWG, 256, 0, stream>>>(Z, T, P);
    reduce_kernel<<<(NBATCH * S_STRIDE + 255) / 256, 256, 0, stream>>>(P, S);
    finalize_kernel<<<1, 256, 0, stream>>>(S, out);
}
